// TransformerXLBlock_19439021981918
// MI455X (gfx1250) — matrix operand order repaired and checked by reading, not compiled, not run on hardware
//
#include <hip/hip_runtime.h>
#include <cstdint>
#include <cstddef>

// ---------------------------------------------------------------------------
// Transformer-XL block for MI455X (gfx1250, wave32, WMMA bf16 16x16x32).
// All GEMMs run on v_wmma_f32_16x16x32_bf16. Attention is flash-style
// (streaming softmax) so the 512MB score tensor never touches HBM.
// GEMM main loops are double-buffered; tile staging uses the CDNA5 async
// global->LDS path (GLOBAL_LOAD_ASYNC_TO_LDS_B128 + s_wait_asynccnt),
// overlapping HBM feed with the matrix pipes.
// ---------------------------------------------------------------------------

typedef __bf16 bf16;
typedef __attribute__((ext_vector_type(16))) __bf16 v16bf;
typedef __attribute__((ext_vector_type(8)))  __bf16 v8bf;
typedef __attribute__((ext_vector_type(8)))  float  v8f;
typedef __attribute__((ext_vector_type(4)))  int    v4i;

#define DEV __device__ __forceinline__

#if defined(__AMDGCN__) && __has_builtin(__builtin_amdgcn_global_load_async_to_lds_b128)
#define HAVE_ASYNC_LDS 1
#else
#define HAVE_ASYNC_LDS 0
#endif

DEV bf16 tobf(float f) {
  unsigned u = __builtin_bit_cast(unsigned, f);
  u += 0x7fffu + ((u >> 16) & 1u);
  unsigned short s = (unsigned short)(u >> 16);
  return __builtin_bit_cast(bf16, s);
}

DEV v16bf cat8(v8bf lo, v8bf hi) {
  return __builtin_shufflevector(lo, hi, 0,1,2,3,4,5,6,7,8,9,10,11,12,13,14,15);
}

// A-fragment (16x32 bf16) from row-major LDS tile.
// ISA layout: lanes 0-15 -> row=lane, K {0..7,16..23}; lanes 16-31 -> row=lane-16, K {8..15,24..31}
DEV v16bf fragA(const bf16* base, int stride, int m_off, int k_off, int lane) {
  int l  = lane & 15;
  int c0 = (lane < 16) ? 0 : 8;
  const bf16* p = base + (size_t)(m_off + l) * stride + k_off + c0;
  v8bf lo = *(const v8bf*)(p);
  v8bf hi = *(const v8bf*)(p + 16);
  return cat8(lo, hi);
}

// B-fragment (32x16 bf16) from n-major (transposed) LDS tile Bt[n][k].
// Lanes 0-15: n=lane, K {0..7,16..23}; lanes 16-31: n=lane-16, K {8..15,24..31}
DEV v16bf fragB(const bf16* base, int stride, int n_off, int k_off, int lane) {
  int l  = lane & 15;
  int c0 = (lane < 16) ? 0 : 8;
  const bf16* p = base + (size_t)(n_off + l) * stride + k_off + c0;
  v8bf lo = *(const v8bf*)(p);
  v8bf hi = *(const v8bf*)(p + 16);
  return cat8(lo, hi);
}

DEV v8f wmma_bf16(v16bf a, v16bf b, v8f c) {
  return __builtin_amdgcn_wmma_f32_16x16x32_bf16(false, a, false, b, (short)0, c, false, false);
}

// Async 16B global -> LDS copy (ASYNCcnt-tracked on CDNA5).
// Builtin signature (from hipcc diagnostic): (v4i AS1*, v4i AS3*, imm, imm)
DEV void async16(const bf16* g, bf16* l) {
#if HAVE_ASYNC_LDS
  __builtin_amdgcn_global_load_async_to_lds_b128(
      (__attribute__((address_space(1))) v4i*)g,
      (__attribute__((address_space(3))) v4i*)l, 0, 0);
#else
  (void)g; (void)l;
#endif
}

DEV void wait_async0() {
#if HAVE_ASYNC_LDS
#if __has_builtin(__builtin_amdgcn_s_wait_asynccnt)
  __builtin_amdgcn_s_wait_asynccnt(0);
#else
  asm volatile("s_wait_asynccnt 0" ::: "memory");
#endif
#endif
}

static const v8f VZERO8 = {0.f,0.f,0.f,0.f,0.f,0.f,0.f,0.f};

// ---------------------------------------------------------------------------
// f32 [K][N] -> bf16 [N][K] transposed convert (LDS tiled)
// ---------------------------------------------------------------------------
__global__ __launch_bounds__(256) void transpose_cvt_kernel(
    const float* __restrict__ in, bf16* __restrict__ out, int K, int N) {
  __shared__ float t[32][33];
  int n0 = blockIdx.x * 32, k0 = blockIdx.y * 32;
  int tx = threadIdx.x, ty = threadIdx.y; // (32,8)
#pragma unroll
  for (int r = 0; r < 4; ++r)
    t[ty + r * 8][tx] = in[(size_t)(k0 + ty + r * 8) * N + n0 + tx];
  __syncthreads();
#pragma unroll
  for (int r = 0; r < 4; ++r)
    out[(size_t)(n0 + ty + r * 8) * K + k0 + tx] = tobf(t[tx][ty + r * 8]);
}

__global__ void cvt_flat_kernel(const float* __restrict__ in, bf16* __restrict__ out, int n) {
  for (int i = blockIdx.x * blockDim.x + threadIdx.x; i < n; i += gridDim.x * blockDim.x)
    out[i] = tobf(in[i]);
}

// mem f32 [4][1024][1024] -> kinput bf16 rows b*2048 + pos
__global__ void scatter_mem_kernel(const float* __restrict__ src, bf16* __restrict__ kin) {
  const int n = 4 * 1024 * 1024;
  for (int i = blockIdx.x * blockDim.x + threadIdx.x; i < n; i += gridDim.x * blockDim.x) {
    int b = i >> 20;
    int rem = i & ((1 << 20) - 1);
    int row = rem >> 10, col = rem & 1023;
    kin[((size_t)(b * 2048 + row)) * 1024 + col] = tobf(src[i]);
  }
}

__global__ void copy_newmem_kernel(float* __restrict__ out) {
  const int n = 4 * 1024 * 1024;
  for (int i = blockIdx.x * blockDim.x + threadIdx.x; i < n; i += gridDim.x * blockDim.x)
    out[(size_t)n + i] = out[i];
}

// ---------------------------------------------------------------------------
// LayerNorm over rows of 1024, emits bf16 (optionally also scattered into the
// concat(mem, x) K-input buffer).
// ---------------------------------------------------------------------------
__global__ __launch_bounds__(256) void ln_kernel(
    const float* __restrict__ x, const float* __restrict__ g, const float* __restrict__ b,
    bf16* __restrict__ dst1, bf16* __restrict__ dst2, int mapDst2) {
  const int row = blockIdx.x;
  const int tid = threadIdx.x;
  const size_t base = (size_t)row * 1024;
  float4 v = reinterpret_cast<const float4*>(x + base)[tid];
  float s  = v.x + v.y + v.z + v.w;
  float sq = v.x * v.x + v.y * v.y + v.z * v.z + v.w * v.w;
#pragma unroll
  for (int o = 16; o > 0; o >>= 1) { s += __shfl_down(s, o); sq += __shfl_down(sq, o); }
  __shared__ float ps[8], pq[8];
  __shared__ float mean_s, rstd_s;
  if ((tid & 31) == 0) { ps[tid >> 5] = s; pq[tid >> 5] = sq; }
  __syncthreads();
  if (tid == 0) {
    float S = 0.f, Q = 0.f;
#pragma unroll
    for (int i = 0; i < 8; ++i) { S += ps[i]; Q += pq[i]; }
    float mean = S * (1.0f / 1024.0f);
    float var  = Q * (1.0f / 1024.0f) - mean * mean;
    mean_s = mean;
    rstd_s = rsqrtf(var + 1e-5f);
  }
  __syncthreads();
  float mean = mean_s, rstd = rstd_s;
  size_t base2 = 0;
  if (mapDst2) {
    int bb = row >> 10, q = row & 1023;
    base2 = ((size_t)(bb * 2048 + 1024 + q)) * 1024;
  }
  float vv[4] = {v.x, v.y, v.z, v.w};
#pragma unroll
  for (int e = 0; e < 4; ++e) {
    int c = tid * 4 + e;
    float y = (vv[e] - mean) * rstd * g[c] + b[c];
    bf16 h = tobf(y);
    dst1[base + c] = h;
    if (mapDst2) dst2[base2 + c] = h;
  }
}

// ---------------------------------------------------------------------------
// WMMA GEMM: C[M,N] = A[M,K] @ B[K,N], A bf16 row-major, B given as Bt[N][K].
// Block: 256 thr = 8 waves, tile 128x128, double-buffered LDS with async
// global->LDS staging. Wave grid 4x2, each wave 32x64.
// OUTMODE: 0 f32 out; 1 bf16 out; 2 dual-bias bf16 (q + r_w_bias / q + r_r_bias);
//          3 bf16 V output transposed per-head: vT[b][h][d][key]
// ---------------------------------------------------------------------------
template<int OUTMODE, bool BIAS, bool DOGELU, bool RES>
__global__ __launch_bounds__(256) void gemm_wmma_kernel(
    const bf16* __restrict__ A, const bf16* __restrict__ Bt,
    float* __restrict__ outF, bf16* __restrict__ outB0, bf16* __restrict__ outB1,
    const float* __restrict__ bias0, const float* __restrict__ bias1,
    const float* __restrict__ res, int M, int N, int K) {
  __shared__ alignas(16) bf16 As[2][128][40];
  __shared__ alignas(16) bf16 Bs[2][128][40];
  const int tid = threadIdx.x;
  const int m0 = blockIdx.y * 128;
  const int n0 = blockIdx.x * 128;
  const int lane = tid & 31;
  const int wid  = tid >> 5;
  const int wm = wid >> 1;  // 0..3
  const int wn = wid & 1;   // 0..1

  v8f acc[2][4];
#pragma unroll
  for (int i = 0; i < 2; ++i)
#pragma unroll
    for (int j = 0; j < 4; ++j) acc[i][j] = VZERO8;

  const int srow = tid >> 1;
  const int scol = (tid & 1) * 16;
  const bf16* gArow = A  + (size_t)(m0 + srow) * K + scol;
  const bf16* gBrow = Bt + (size_t)(n0 + srow) * K + scol;

  auto issue_stage = [&](int kt) {
    const int k0 = kt * 32;
    const bf16* ga = gArow + k0;
    const bf16* gb = gBrow + k0;
    bf16* la = &As[kt & 1][srow][scol];
    bf16* lb = &Bs[kt & 1][srow][scol];
#if HAVE_ASYNC_LDS
    async16(ga,     la);
    async16(ga + 8, la + 8);
    async16(gb,     lb);
    async16(gb + 8, lb + 8);
#else
    const v8bf* gav = (const v8bf*)ga;
    const v8bf* gbv = (const v8bf*)gb;
    *(v8bf*)la       = gav[0];
    *(v8bf*)(la + 8) = gav[1];
    *(v8bf*)lb       = gbv[0];
    *(v8bf*)(lb + 8) = gbv[1];
#endif
  };

  const int nk = K >> 5;
  issue_stage(0);
  wait_async0();
  __syncthreads();

  for (int kt = 0; kt < nk; ++kt) {
    if (kt + 1 < nk) {
      issue_stage(kt + 1);
#if !HAVE_ASYNC_LDS
      if (kt + 2 < nk) {  // software prefetch of the tile after next
        __builtin_prefetch(gArow + (kt + 2) * 32, 0, 0);
        __builtin_prefetch(gBrow + (kt + 2) * 32, 0, 0);
      }
#endif
    }
    const bf16* asb = &As[kt & 1][0][0];
    const bf16* bsb = &Bs[kt & 1][0][0];
    v16bf af[2], bfr[4];
#pragma unroll
    for (int sm = 0; sm < 2; ++sm) af[sm] = fragA(asb, 40, wm * 32 + sm * 16, 0, lane);
#pragma unroll
    for (int sn = 0; sn < 4; ++sn) bfr[sn] = fragB(bsb, 40, wn * 64 + sn * 16, 0, lane);
#pragma unroll
    for (int sm = 0; sm < 2; ++sm)
#pragma unroll
      for (int sn = 0; sn < 4; ++sn)
        acc[sm][sn] = wmma_bf16(af[sm], bfr[sn], acc[sm][sn]);
    wait_async0();
    __syncthreads();
  }

  const int mbase = (lane & 16) ? 8 : 0;
#pragma unroll
  for (int sm = 0; sm < 2; ++sm)
#pragma unroll
    for (int sn = 0; sn < 4; ++sn)
#pragma unroll
      for (int gi = 0; gi < 8; ++gi) {
        int r = m0 + wm * 32 + sm * 16 + mbase + gi;
        int c = n0 + wn * 64 + sn * 16 + (lane & 15);
        float v = acc[sm][sn][gi];
        if (BIAS)   v += bias0[c];
        if (DOGELU) v = 0.5f * v * (1.0f + erff(v * 0.70710678118f));
        if (RES)    v += res[(size_t)r * N + c];
        if (OUTMODE == 0) {
          outF[(size_t)r * N + c] = v;
        } else if (OUTMODE == 1) {
          outB0[(size_t)r * N + c] = tobf(v);
        } else if (OUTMODE == 2) {
          outB0[(size_t)r * N + c] = tobf(v + bias0[c]);
          outB1[(size_t)r * N + c] = tobf(v + bias1[c]);
        } else {
          int h = c >> 6, d = c & 63;
          int bb = r >> 11, key = r & 2047;
          outB0[((size_t)((bb * 16 + h) * 64 + d)) * 2048 + key] = tobf(v);
        }
      }
}

// ---------------------------------------------------------------------------
// Flash attention with Transformer-XL relative bias.
// Block = 64 threads (2 waves), each wave owns 16 query rows (32 per block).
// k-tiles of 64 keys. BD rel-shift done via LDS gather from a 96-wide r band:
// score[i][j] = AC[i][j] + BDraw[i][ j - i + Q - 1 ].
// ---------------------------------------------------------------------------
__global__ __launch_bounds__(64) void flash_attn_kernel(
    const bf16* __restrict__ qrw, const bf16* __restrict__ qrr,
    const bf16* __restrict__ kmat, const bf16* __restrict__ vT,
    const bf16* __restrict__ rmat, bf16* __restrict__ avec) {
  __shared__ alignas(16) bf16 Qs[32][72];
  __shared__ alignas(16) bf16 Kt[64][72];
  __shared__ alignas(16) bf16 Vt[64][72];
  __shared__ alignas(16) bf16 Rb[96][72];
  __shared__ alignas(16) bf16 Ps[2][16][72];
  __shared__ float Sx[2][16][68];
  __shared__ float BDs[2][16][84];
  __shared__ float rowAlpha[2][16];
  __shared__ float rowLinv[2][16];

  const int tid = threadIdx.x;
  const int w = tid >> 5, lane = tid & 31;
  const int b = blockIdx.z, h = blockIdx.y;
  const int i0 = blockIdx.x * 32;
  const int row = lane & 15, half = lane >> 4;
  const int irow = i0 + 16 * w + row;

  // Stage q (+r_w_bias) then q (+r_r_bias), extract A-fragments.
  v16bf aQw[2], aQr[2];
  for (int pass = 0; pass < 2; ++pass) {
    const bf16* src = pass ? qrr : qrw;
#pragma unroll
    for (int it = 0; it < 4; ++it) {
      int lin = tid + it * 64;
      int rr = lin >> 3, cc = (lin & 7) * 8;
      *(v8bf*)&Qs[rr][cc] =
          *(const v8bf*)(src + (size_t)(b * 1024 + i0 + rr) * 1024 + h * 64 + cc);
    }
    __syncthreads();
#pragma unroll
    for (int fi = 0; fi < 2; ++fi) {
      v16bf f = fragA(&Qs[0][0], 72, 16 * w, fi * 32, lane);
      if (pass) aQr[fi] = f; else aQw[fi] = f;
    }
    __syncthreads();
  }

  v8f o[4];
#pragma unroll
  for (int dt = 0; dt < 4; ++dt) o[dt] = VZERO8;
  float mrow = -3.0e38f, lrow = 0.0f;

  int jmaxi = i0 + 31 + 1024;
  if (jmaxi > 2047) jmaxi = 2047;

  for (int j0 = 0; j0 <= jmaxi; j0 += 64) {
    // ---- stage K tile (rows = keys, cols = d) ----
#pragma unroll
    for (int it = 0; it < 8; ++it) {
      int lin = tid + it * 64;
      int jj = lin >> 3, cc = (lin & 7) * 8;
      union { v8bf v; unsigned u[4]; } tmp;
      if (j0 + jj < 2048)
        tmp.v = *(const v8bf*)(kmat + (size_t)(b * 2048 + j0 + jj) * 1024 + h * 64 + cc);
      else { tmp.u[0] = tmp.u[1] = tmp.u[2] = tmp.u[3] = 0u; }
      *(v8bf*)&Kt[jj][cc] = tmp.v;
    }
    // ---- stage V^T tile (rows = d, cols = keys) ----
#pragma unroll
    for (int it = 0; it < 8; ++it) {
      int lin = tid + it * 64;
      int dd = lin >> 3, cc = (lin & 7) * 8;
      union { v8bf v; unsigned u[4]; } tmp;
      if (j0 + cc < 2048)
        tmp.v = *(const v8bf*)(vT + ((size_t)((b * 16 + h) * 64 + dd)) * 2048 + j0 + cc);
      else { tmp.u[0] = tmp.u[1] = tmp.u[2] = tmp.u[3] = 0u; }
      *(v8bf*)&Vt[dd][cc] = tmp.v;
    }
    // ---- stage relative-position band: d = j - i + 1023 ----
    const int dbase = j0 - i0 + 992;
#pragma unroll
    for (int it = 0; it < 12; ++it) {
      int lin = tid + it * 64;
      int t = lin >> 3, cc = (lin & 7) * 8;
      int dg = dbase + t;
      dg = dg < 0 ? 0 : (dg > 2047 ? 2047 : dg);
      *(v8bf*)&Rb[t][cc] = *(const v8bf*)(rmat + (size_t)dg * 1024 + h * 64 + cc);
    }
    __syncthreads();

    // ---- AC = (q + r_w_bias) @ K^T ----
#pragma unroll
    for (int nt = 0; nt < 4; ++nt) {
      v8f acc = VZERO8;
#pragma unroll
      for (int fi = 0; fi < 2; ++fi)
        acc = wmma_bf16(aQw[fi], fragB(&Kt[0][0], 72, nt * 16, fi * 32, lane), acc);
#pragma unroll
      for (int gi = 0; gi < 8; ++gi)
        Sx[w][((lane & 16) ? 8 : 0) + gi][nt * 16 + (lane & 15)] = acc[gi];
    }
    // ---- BD band = (q + r_r_bias) @ r^T (wave band starts at Rb row 16*(1-w)) ----
    const int t0 = 16 * (1 - w);
#pragma unroll
    for (int bt = 0; bt < 5; ++bt) {
      v8f acc = VZERO8;
#pragma unroll
      for (int fi = 0; fi < 2; ++fi)
        acc = wmma_bf16(aQr[fi], fragB(&Rb[0][0], 72, t0 + bt * 16, fi * 32, lane), acc);
#pragma unroll
      for (int gi = 0; gi < 8; ++gi)
        BDs[w][((lane & 16) ? 8 : 0) + gi][bt * 16 + (lane & 15)] = acc[gi];
    }

    // ---- online softmax: 2 lanes per row, 32 cols each ----
    float lmax = -3.0e38f;
#pragma unroll
    for (int c = 0; c < 32; ++c) {
      int jl = half * 32 + c;
      float s = Sx[w][row][jl] + BDs[w][row][jl - row + 15];
      s *= 0.125f;  // 1/sqrt(64)
      int jg = j0 + jl;
      if (jg > irow + 1024 || jg >= 2048) s = -3.0e38f;
      Sx[w][row][jl] = s;
      lmax = fmaxf(lmax, s);
    }
    lmax = fmaxf(lmax, __shfl_xor(lmax, 16));
    float mnew  = fmaxf(mrow, lmax);
    float alpha = __expf(mrow - mnew);
    float lsum  = 0.0f;
#pragma unroll
    for (int c = 0; c < 32; ++c) {
      int jl = half * 32 + c;
      float p = __expf(Sx[w][row][jl] - mnew);
      lsum += p;
      Ps[w][row][jl] = tobf(p);
    }
    lsum += __shfl_xor(lsum, 16);
    lrow = lrow * alpha + lsum;
    mrow = mnew;
    if (half == 0) rowAlpha[w][row] = alpha;

    // ---- rescale output accumulators ----
    const int mb = (lane & 16) ? 8 : 0;
#pragma unroll
    for (int dt = 0; dt < 4; ++dt)
#pragma unroll
      for (int gi = 0; gi < 8; ++gi)
        o[dt][gi] *= rowAlpha[w][mb + gi];

    // ---- O += P @ V ----
#pragma unroll
    for (int dt = 0; dt < 4; ++dt)
#pragma unroll
      for (int fi = 0; fi < 2; ++fi)
        o[dt] = wmma_bf16(fragA(&Ps[w][0][0], 72, 0, fi * 32, lane),
                          fragB(&Vt[0][0], 72, dt * 16, fi * 32, lane), o[dt]);
    __syncthreads();
  }

  if (half == 0) rowLinv[w][row] = 1.0f / lrow;
  __syncthreads();
  const int mb = (lane & 16) ? 8 : 0;
#pragma unroll
  for (int dt = 0; dt < 4; ++dt)
#pragma unroll
    for (int gi = 0; gi < 8; ++gi) {
      int m  = mb + gi;
      int ir = i0 + 16 * w + m;
      int d  = dt * 16 + (lane & 15);
      avec[(size_t)(b * 1024 + ir) * 1024 + h * 64 + d] = tobf(o[dt][gi] * rowLinv[w][m]);
    }
}

// ---------------------------------------------------------------------------
// Host launcher
// ---------------------------------------------------------------------------
extern "C" void kernel_launch(void* const* d_in, const int* in_sizes, int n_in,
                              void* d_out, int out_size, void* d_ws, size_t ws_size,
                              hipStream_t stream) {
  (void)in_sizes; (void)n_in; (void)out_size; (void)ws_size;
  const float* x    = (const float*)d_in[0];
  const float* pos  = (const float*)d_in[1];
  const float* mem  = (const float*)d_in[3];
  const float* Wq   = (const float*)d_in[4];
  const float* Wk   = (const float*)d_in[5];
  const float* Wv   = (const float*)d_in[6];
  const float* Wr   = (const float*)d_in[7];
  const float* Wo   = (const float*)d_in[8];
  const float* rwb  = (const float*)d_in[9];
  const float* rrb  = (const float*)d_in[10];
  const float* fc1w = (const float*)d_in[11];
  const float* fc1b = (const float*)d_in[12];
  const float* fc2w = (const float*)d_in[13];
  const float* fc2b = (const float*)d_in[14];
  const float* ln1g = (const float*)d_in[15];
  const float* ln1b = (const float*)d_in[16];
  const float* ln2g = (const float*)d_in[17];
  const float* ln2b = (const float*)d_in[18];
  float* out = (float*)d_out;

  char* ws = (char*)d_ws;
  size_t off = 0;
  auto alloc = [&](size_t bytes) -> void* {
    void* p = ws + off;
    off = (off + bytes + 255) & ~(size_t)255;
    return p;
  };
  const size_t MB = 1024 * 1024;
  bf16* WqT  = (bf16*)alloc(2 * MB);
  bf16* WkT  = (bf16*)alloc(2 * MB);
  bf16* WvT  = (bf16*)alloc(2 * MB);
  bf16* WrT  = (bf16*)alloc(2 * MB);
  bf16* WoT  = (bf16*)alloc(2 * MB);
  bf16* fc1T = (bf16*)alloc(8 * MB);
  bf16* fc2T = (bf16*)alloc(8 * MB);
  bf16* posb = (bf16*)alloc(4 * MB);
  bf16* ain  = (bf16*)alloc(8 * MB);   // layernorm(x) bf16 [4096][1024]
  bf16* kin  = (bf16*)alloc(16 * MB);  // concat(mem, ain)   [8192][1024]
  bf16* qrw  = (bf16*)alloc(8 * MB);
  bf16* qrr  = (bf16*)alloc(8 * MB);
  bf16* kb   = (bf16*)alloc(16 * MB);  // K proj [8192][1024]
  bf16* vT   = (bf16*)alloc(16 * MB);  // V proj transposed [b][h][d][key]
  bf16* rb   = (bf16*)alloc(4 * MB);   // R proj [2048][1024]
  bf16* avec = (bf16*)alloc(8 * MB);   // attention output [4096][1024]
  float* x2  = (float*)alloc(16 * MB); // x + attn_out
  bf16* h1   = (bf16*)alloc(8 * MB);   // layernorm2(x2)
  bf16* mid  = (bf16*)alloc(32 * MB);  // FFN hidden [4096][4096]

  dim3 tb(32, 8);
  // weight transpose+convert
  transpose_cvt_kernel<<<dim3(32, 32),  tb, 0, stream>>>(Wq,   WqT,  1024, 1024);
  transpose_cvt_kernel<<<dim3(32, 32),  tb, 0, stream>>>(Wk,   WkT,  1024, 1024);
  transpose_cvt_kernel<<<dim3(32, 32),  tb, 0, stream>>>(Wv,   WvT,  1024, 1024);
  transpose_cvt_kernel<<<dim3(32, 32),  tb, 0, stream>>>(Wr,   WrT,  1024, 1024);
  transpose_cvt_kernel<<<dim3(32, 32),  tb, 0, stream>>>(Wo,   WoT,  1024, 1024);
  transpose_cvt_kernel<<<dim3(128, 32), tb, 0, stream>>>(fc1w, fc1T, 1024, 4096);
  transpose_cvt_kernel<<<dim3(32, 128), tb, 0, stream>>>(fc2w, fc2T, 4096, 1024);
  cvt_flat_kernel<<<2048, 256, 0, stream>>>(pos, posb, 2048 * 1024);
  scatter_mem_kernel<<<2048, 256, 0, stream>>>(mem, kin);

  // LN1 -> ain, also scattered into kin rows [b*2048+1024 ..]
  ln_kernel<<<4096, 256, 0, stream>>>(x, ln1g, ln1b, ain, kin, 1);

  // Q projection with fused dual relative-bias outputs
  gemm_wmma_kernel<2, false, false, false><<<dim3(8, 32), 256, 0, stream>>>(
      ain, WqT, nullptr, qrw, qrr, rwb, rrb, nullptr, 4096, 1024, 1024);
  // K projection
  gemm_wmma_kernel<1, false, false, false><<<dim3(8, 64), 256, 0, stream>>>(
      kin, WkT, nullptr, kb, nullptr, nullptr, nullptr, nullptr, 8192, 1024, 1024);
  // V projection (per-head transposed output)
  gemm_wmma_kernel<3, false, false, false><<<dim3(8, 64), 256, 0, stream>>>(
      kin, WvT, nullptr, vT, nullptr, nullptr, nullptr, nullptr, 8192, 1024, 1024);
  // R projection
  gemm_wmma_kernel<1, false, false, false><<<dim3(8, 16), 256, 0, stream>>>(
      posb, WrT, nullptr, rb, nullptr, nullptr, nullptr, nullptr, 2048, 1024, 1024);

  // flash attention: 32 q-rows per block, 16 heads, 4 batches
  flash_attn_kernel<<<dim3(32, 16, 4), 64, 0, stream>>>(qrw, qrr, kb, vT, rb, avec);

  // output projection + residual -> x2 (f32)
  gemm_wmma_kernel<0, false, false, true><<<dim3(8, 32), 256, 0, stream>>>(
      avec, WoT, x2, nullptr, nullptr, nullptr, nullptr, x, 4096, 1024, 1024);

  // LN2
  ln_kernel<<<4096, 256, 0, stream>>>(x2, ln2g, ln2b, h1, nullptr, 0);

  // FFN1: bias + exact GELU
  gemm_wmma_kernel<1, true, true, false><<<dim3(32, 32), 256, 0, stream>>>(
      h1, fc1T, nullptr, mid, nullptr, fc1b, nullptr, nullptr, 4096, 4096, 1024);
  // FFN2: bias + residual -> out (f32)
  gemm_wmma_kernel<0, true, false, true><<<dim3(8, 32), 256, 0, stream>>>(
      mid, fc2T, out, nullptr, nullptr, fc2b, nullptr, x2, 4096, 1024, 4096);

  // new_mem = concat(mem,out)[-1024:] == out (QLEN == MEM_LEN)
  copy_newmem_kernel<<<2048, 256, 0, stream>>>(out);
}
